// DeepSeekMoE_47425028882761
// MI455X (gfx1250) — compile-verified, matrix operand order checked by reading, not run on hardware
//
#include <hip/hip_runtime.h>
#include <hip/hip_bf16.h>
#include <math.h>

// Problem constants (match reference)
#define Dn  2048
#define Fn  1408
#define En  8
#define FSn 2816   // F * N_SHARED
#define Nn  2048   // 2 * 1024 tokens

typedef __attribute__((ext_vector_type(16))) __bf16 v16bf;
typedef __attribute__((ext_vector_type(8)))  __bf16 v8bf;
typedef __attribute__((ext_vector_type(4)))  __bf16 v4bf;
typedef __attribute__((ext_vector_type(8)))  float  v8f;

// ---------------------------------------------------------------------------
// Swizzled LDS tile:  tile[row][k], row stride 32 bf16 (64 B = 4 chunks of 16B).
// Physical chunk index = logicalChunk ^ (row & 3)  -> bank-conflict-free for
// both the staging stores and the 16B fragment loads.
//
// CDNA5 16-bit A/B fragment (wave32): lane = row (mod 16); halves 0..7 hold
// K = lhalf*8..+7 (logical chunks 0/1), halves 8..15 hold K+16 (chunks 2/3).
// ---------------------------------------------------------------------------
__device__ __forceinline__ v16bf frag_ld(const __bf16* tile, int row, int lhalf) {
  const __bf16* rp = tile + row * 32;
  const int sw = row & 3;
  const v8bf lo = *(const v8bf*)(rp + ((lhalf)     ^ sw) * 8);
  const v8bf hi = *(const v8bf*)(rp + ((2 + lhalf) ^ sw) * 8);
  return __builtin_shufflevector(lo, hi, 0,1,2,3,4,5,6,7,8,9,10,11,12,13,14,15);
}

__device__ __forceinline__ v8f wmma_bf16(v16bf a, v16bf b, v8f c) {
  return __builtin_amdgcn_wmma_f32_16x16x32_bf16(false, a, false, b, (short)0, c,
                                                 false, false);
}

__device__ __forceinline__ float fast_silu(float x) {
  return x * __builtin_amdgcn_rcpf(1.f + __expf(-x));
}

// ---------------------------------------------------------------------------
// 1) x fp32 -> bf16 (x is re-read by 9 up-projections; halve its bandwidth)
// ---------------------------------------------------------------------------
__global__ __launch_bounds__(256)
void moe_cvt_bf16_kernel(const float* __restrict__ in, __bf16* __restrict__ out,
                         int n4) {
  int i = blockIdx.x * blockDim.x + threadIdx.x;
  if (i >= n4) return;
  float4 v = ((const float4*)in)[i];
  v4bf o;
  o[0] = (__bf16)v.x; o[1] = (__bf16)v.y; o[2] = (__bf16)v.z; o[3] = (__bf16)v.w;
  ((v4bf*)out)[i] = o;
}

// ---------------------------------------------------------------------------
// 2) Gate: one wave32 per token. logits = x@gate_w, softmax, top-2 -> cw[N,E]
// ---------------------------------------------------------------------------
__global__ __launch_bounds__(256)
void moe_gate_kernel(const float* __restrict__ x, const float* __restrict__ gw,
                     float* __restrict__ cw) {
  const int wave = threadIdx.x >> 5;
  const int lane = threadIdx.x & 31;
  const int tok  = blockIdx.x * 8 + wave;

  float acc[En];
#pragma unroll
  for (int e = 0; e < En; ++e) acc[e] = 0.f;

  const float* xr = x + (size_t)tok * Dn;
  for (int d = lane; d < Dn; d += 32) {
    const float xv = xr[d];
#pragma unroll
    for (int e = 0; e < En; ++e) acc[e] += xv * gw[d * En + e];
  }
#pragma unroll
  for (int off = 16; off >= 1; off >>= 1) {
#pragma unroll
    for (int e = 0; e < En; ++e) acc[e] += __shfl_down(acc[e], off, 32);
  }
  if (lane == 0) {
    float m = acc[0];
#pragma unroll
    for (int e = 1; e < En; ++e) m = fmaxf(m, acc[e]);
    float p[En]; float s = 0.f;
#pragma unroll
    for (int e = 0; e < En; ++e) { p[e] = __expf(acc[e] - m); s += p[e]; }
    const float inv = 1.f / s;
#pragma unroll
    for (int e = 0; e < En; ++e) p[e] *= inv;
    int i1 = 0;
#pragma unroll
    for (int e = 1; e < En; ++e) if (p[e] > p[i1]) i1 = e;
    int i2 = (i1 == 0) ? 1 : 0;
#pragma unroll
    for (int e = 0; e < En; ++e) if (e != i1 && p[e] > p[i2]) i2 = e;
#pragma unroll
    for (int e = 0; e < En; ++e)
      cw[(size_t)tok * En + e] = (e == i1) ? p[i1] : ((e == i2) ? p[i2] : 0.f);
  }
}

// ---------------------------------------------------------------------------
// 3) Up-projection + SwiGLU:  H = silu(Xb@W1) * (Xb@W3), H bf16 [N, FOUT]
//    Block tile 128(M) x 64(N), BK=32, 8 waves x (32x32), dual accumulators.
//    Double-buffered LDS + register prefetch: one barrier per K-step.
//    FOUT is a template constant so all weight-load offsets are immediates.
// ---------------------------------------------------------------------------
template <int FOUT>
__global__ __launch_bounds__(256)
void moe_up_swiglu_kernel(const __bf16* __restrict__ Xb,  // [N, D] bf16
                          const float*  __restrict__ W1,  // [D, FOUT] fp32
                          const float*  __restrict__ W3,  // [D, FOUT] fp32
                          __bf16*       __restrict__ H) { // [N, FOUT] bf16
  __shared__ __align__(16) __bf16 sA [2][128 * 32];  // 16 KB
  __shared__ __align__(16) __bf16 sB1[2][64 * 32];   //  8 KB (transposed [n][k])
  __shared__ __align__(16) __bf16 sB3[2][64 * 32];   //  8 KB

  const int tid   = threadIdx.x;
  const int lane  = tid & 31;
  const int lrow  = lane & 15;
  const int lhalf = lane >> 4;
  const int wave  = tid >> 5;
  const int mw    = (wave & 3) << 5;   // wave M offset (0,32,64,96)
  const int nw    = (wave >> 2) << 5;  // wave N offset (0,32)

  const int m0 = blockIdx.x * 128;
  const int n0 = blockIdx.y * 64;

  // A staging: each thread copies 32 B (two 16-B chunks) of one row
  const int arow = tid >> 1;            // 0..127
  const int acp  = (tid & 1) << 1;      // logical chunk pair: 0 or 2
  // B staging: 2 tasks/thread/matrix; task r = (n = tid&63, g = tid>>6 + 4r)
  const int bn  = tid & 63;
  const int bg0 = tid >> 6;             // k-group 0..3 (4 k each)

  // Loop-carried pointers; all per-load offsets below are literal immediates.
  const __bf16* pA = Xb + (size_t)(m0 + arow) * Dn + acp * 8;
  const float*  p1 = W1 + (size_t)(bg0 * 4) * FOUT + n0 + bn;
  const float*  p3 = W3 + (size_t)(bg0 * 4) * FOUT + n0 + bn;

  uint4 aR0, aR1;
  float b1R[2][4], b3R[2][4];

  v8f acc1[2][2] = {};
  v8f acc3[2][2] = {};

  const int KS = Dn / 32;
  // ---- prefetch K-step 0 into registers
  aR0 = *(const uint4*)(pA);
  aR1 = *(const uint4*)(pA + 8);
#pragma unroll
  for (int r = 0; r < 2; ++r)
#pragma unroll
    for (int j = 0; j < 4; ++j) {
      b1R[r][j] = p1[(size_t)(r * 16 + j) * FOUT];
      b3R[r][j] = p3[(size_t)(r * 16 + j) * FOUT];
    }

  for (int it = 0; it < KS; ++it) {
    const int buf = it & 1;
    // ---- stage registers -> LDS (swizzled)
    {
      __bf16* A = sA[buf];
      const int sw = arow & 3;
      *(uint4*)(A + arow * 32 + ((acp)     ^ sw) * 8) = aR0;
      *(uint4*)(A + arow * 32 + ((acp + 1) ^ sw) * 8) = aR1;
#pragma unroll
      for (int r = 0; r < 2; ++r) {
        const int g = bg0 + r * 4;
        v4bf q1, q3;
#pragma unroll
        for (int j = 0; j < 4; ++j) {
          q1[j] = (__bf16)b1R[r][j];
          q3[j] = (__bf16)b3R[r][j];
        }
        const int off = bn * 32 + (((g >> 1) ^ (bn & 3)) * 8) + ((g & 1) * 4);
        *(v4bf*)(sB1[buf] + off) = q1;
        *(v4bf*)(sB3[buf] + off) = q3;
      }
    }
    __syncthreads();
    // ---- prefetch next K-step while this one computes
    if (it + 1 < KS) {
      pA += 32;
      p1 += (size_t)32 * FOUT;
      p3 += (size_t)32 * FOUT;
      aR0 = *(const uint4*)(pA);
      aR1 = *(const uint4*)(pA + 8);
#pragma unroll
      for (int r = 0; r < 2; ++r)
#pragma unroll
        for (int j = 0; j < 4; ++j) {
          b1R[r][j] = p1[(size_t)(r * 16 + j) * FOUT];
          b3R[r][j] = p3[(size_t)(r * 16 + j) * FOUT];
        }
    }
    // ---- fragments + WMMA
    v16bf af[2], b1f[2], b3f[2];
#pragma unroll
    for (int mf = 0; mf < 2; ++mf)
      af[mf] = frag_ld(sA[buf], mw + (mf << 4) + lrow, lhalf);
#pragma unroll
    for (int nf = 0; nf < 2; ++nf) {
      b1f[nf] = frag_ld(sB1[buf], nw + (nf << 4) + lrow, lhalf);
      b3f[nf] = frag_ld(sB3[buf], nw + (nf << 4) + lrow, lhalf);
    }
#pragma unroll
    for (int mf = 0; mf < 2; ++mf)
#pragma unroll
      for (int nf = 0; nf < 2; ++nf) {
        acc1[mf][nf] = wmma_bf16(af[mf], b1f[nf], acc1[mf][nf]);
        acc3[mf][nf] = wmma_bf16(af[mf], b3f[nf], acc3[mf][nf]);
      }
  }

  // Epilogue: silu(h1)*h3, bf16 store. D-matrix: VGPR i -> M=i(+8 for hi lanes)
#pragma unroll
  for (int mf = 0; mf < 2; ++mf)
#pragma unroll
    for (int nf = 0; nf < 2; ++nf) {
      const int mBase = m0 + mw + (mf << 4) + (lhalf << 3);
      const int n     = n0 + nw + (nf << 4) + lrow;
#pragma unroll
      for (int i = 0; i < 8; ++i) {
        const float h1 = acc1[mf][nf][i];
        const float h3 = acc3[mf][nf][i];
        H[(size_t)(mBase + i) * FOUT + n] = (__bf16)(fast_silu(h1) * h3);
      }
    }
}

// ---------------------------------------------------------------------------
// 4) Down-projection:  Y (+)= rowscale * (H @ W2), Y fp32 [N, D]
//    ACCUM==0: Y = acc  (shared-expert pass, initializes Y)
//    ACCUM==1: Y += cw[m,expert] * acc  (routed experts; zero weight => 0)
// ---------------------------------------------------------------------------
template <int FIN, int ACCUM>
__global__ __launch_bounds__(256)
void moe_down_kernel(const __bf16* __restrict__ Hin,  // [N, FIN] bf16
                     const float*  __restrict__ W2,   // [FIN, D] fp32
                     float*        __restrict__ Y,    // [N, D] fp32
                     const float*  __restrict__ cw,   // [N, E] or null
                     int expert) {
  __shared__ __align__(16) __bf16 sA[2][128 * 32];  // 16 KB
  __shared__ __align__(16) __bf16 sB[2][64 * 32];   //  8 KB

  const int tid   = threadIdx.x;
  const int lane  = tid & 31;
  const int lrow  = lane & 15;
  const int lhalf = lane >> 4;
  const int wave  = tid >> 5;
  const int mw    = (wave & 3) << 5;
  const int nw    = (wave >> 2) << 5;

  const int m0 = blockIdx.x * 128;
  const int n0 = blockIdx.y * 64;

  const int arow = tid >> 1;
  const int acp  = (tid & 1) << 1;
  const int bn   = tid & 63;
  const int bg0  = tid >> 6;

  const __bf16* pA = Hin + (size_t)(m0 + arow) * FIN + acp * 8;
  const float*  pB = W2 + (size_t)(bg0 * 4) * Dn + n0 + bn;

  uint4 aR0, aR1;
  float bR[2][4];

  v8f acc[2][2] = {};

  const int KS = FIN / 32;
  aR0 = *(const uint4*)(pA);
  aR1 = *(const uint4*)(pA + 8);
#pragma unroll
  for (int r = 0; r < 2; ++r)
#pragma unroll
    for (int j = 0; j < 4; ++j) bR[r][j] = pB[(size_t)(r * 16 + j) * Dn];

  for (int it = 0; it < KS; ++it) {
    const int buf = it & 1;
    {
      __bf16* A = sA[buf];
      const int sw = arow & 3;
      *(uint4*)(A + arow * 32 + ((acp)     ^ sw) * 8) = aR0;
      *(uint4*)(A + arow * 32 + ((acp + 1) ^ sw) * 8) = aR1;
#pragma unroll
      for (int r = 0; r < 2; ++r) {
        const int g = bg0 + r * 4;
        v4bf pk;
#pragma unroll
        for (int j = 0; j < 4; ++j) pk[j] = (__bf16)bR[r][j];
        const int off = bn * 32 + (((g >> 1) ^ (bn & 3)) * 8) + ((g & 1) * 4);
        *(v4bf*)(sB[buf] + off) = pk;
      }
    }
    __syncthreads();
    if (it + 1 < KS) {
      pA += 32;
      pB += (size_t)32 * Dn;
      aR0 = *(const uint4*)(pA);
      aR1 = *(const uint4*)(pA + 8);
#pragma unroll
      for (int r = 0; r < 2; ++r)
#pragma unroll
        for (int j = 0; j < 4; ++j) bR[r][j] = pB[(size_t)(r * 16 + j) * Dn];
    }
    v16bf af[2], bf[2];
#pragma unroll
    for (int mf = 0; mf < 2; ++mf)
      af[mf] = frag_ld(sA[buf], mw + (mf << 4) + lrow, lhalf);
#pragma unroll
    for (int nf = 0; nf < 2; ++nf)
      bf[nf] = frag_ld(sB[buf], nw + (nf << 4) + lrow, lhalf);
#pragma unroll
    for (int mf = 0; mf < 2; ++mf)
#pragma unroll
      for (int nf = 0; nf < 2; ++nf)
        acc[mf][nf] = wmma_bf16(af[mf], bf[nf], acc[mf][nf]);
  }

#pragma unroll
  for (int mf = 0; mf < 2; ++mf)
#pragma unroll
    for (int nf = 0; nf < 2; ++nf) {
      const int mBase = m0 + mw + (mf << 4) + (lhalf << 3);
      const int n     = n0 + nw + (nf << 4) + lrow;
#pragma unroll
      for (int i = 0; i < 8; ++i) {
        const int m = mBase + i;
        const size_t off = (size_t)m * Dn + n;
        const float v = acc[mf][nf][i];
        if (ACCUM)
          Y[off] += cw[(size_t)m * En + expert] * v;
        else
          Y[off] = v;
      }
    }
}

// ---------------------------------------------------------------------------
// Host launcher. Workspace layout (needs ~20 MB):
//   [0)            Xb  : N*D bf16          (8 MiB)
//   [8MiB)         cw  : N*E fp32          (64 KiB)
//   [8MiB+64KiB)   H   : N*FSn bf16        (11 MiB, reused per expert)
// ---------------------------------------------------------------------------
extern "C" void kernel_launch(void* const* d_in, const int* in_sizes, int n_in,
                              void* d_out, int out_size, void* d_ws,
                              size_t ws_size, hipStream_t stream) {
  (void)in_sizes; (void)n_in; (void)out_size; (void)ws_size;
  const float* x      = (const float*)d_in[0];
  const float* gate_w = (const float*)d_in[1];
  const float* sw1    = (const float*)d_in[2];
  const float* sw2    = (const float*)d_in[3];
  const float* sw3    = (const float*)d_in[4];
  const float* rw1    = (const float*)d_in[5];
  const float* rw2    = (const float*)d_in[6];
  const float* rw3    = (const float*)d_in[7];
  float* Y = (float*)d_out;

  char* ws = (char*)d_ws;
  __bf16* Xb = (__bf16*)ws;
  float*  CW = (float*)(ws + (size_t)Nn * Dn * 2);
  __bf16* H  = (__bf16*)(ws + (size_t)Nn * Dn * 2 + (size_t)Nn * En * 4);

  // 1) x -> bf16
  {
    const int n4 = (Nn * Dn) / 4;
    moe_cvt_bf16_kernel<<<(n4 + 255) / 256, 256, 0, stream>>>(x, Xb, n4);
  }
  // 2) gate / top-2 combine weights
  moe_gate_kernel<<<Nn / 8, 256, 0, stream>>>(x, gate_w, CW);

  // 3) shared expert: initializes Y
  moe_up_swiglu_kernel<FSn><<<dim3(Nn / 128, FSn / 64), 256, 0, stream>>>(
      Xb, sw1, sw3, H);
  moe_down_kernel<FSn, 0><<<dim3(Nn / 128, Dn / 64), 256, 0, stream>>>(
      H, sw2, Y, nullptr, 0);

  // 4) routed experts: accumulate with per-token combine weights
  for (int e = 0; e < En; ++e) {
    moe_up_swiglu_kernel<Fn><<<dim3(Nn / 128, Fn / 64), 256, 0, stream>>>(
        Xb, rw1 + (size_t)e * Dn * Fn, rw3 + (size_t)e * Dn * Fn, H);
    moe_down_kernel<Fn, 1><<<dim3(Nn / 128, Dn / 64), 256, 0, stream>>>(
        H, rw2 + (size_t)e * Fn * Dn, Y, CW, e);
  }
}
